// PostBlock_80676665688528
// MI455X (gfx1250) — 8-bit matrix operand order repaired and checked by reading, compiled, not run on hardware
//
#include <hip/hip_runtime.h>
#include <stdint.h>

#define MPIX  100352
#define CCH   256
#define HWSZ  3136
#define NIMG  32
#define EPSBN 1e-5f

typedef int   v16i __attribute__((ext_vector_type(16)));
typedef float v8f  __attribute__((ext_vector_type(8)));

__device__ __forceinline__ int8_t sgnb(float v) {
  return (int8_t)((v > 0.0f) ? 1 : ((v < 0.0f) ? -1 : 0));
}
__device__ __forceinline__ uint8_t fp8s(float v) {
  return (v > 0.0f) ? (uint8_t)0x38 : ((v < 0.0f) ? (uint8_t)0xB8 : (uint8_t)0x00);
}

__global__ __launch_bounds__(256) void k_init(
    const float* __restrict__ w1, const float* __restrict__ w2,
    const float* __restrict__ w3, const float* __restrict__ wdw,
    uint8_t* __restrict__ sw, int8_t* __restrict__ swdw) {
  int g = blockIdx.x * 256 + threadIdx.x;
  if (g < 2304) swdw[g] = sgnb(wdw[g]);
  sw[g]          = fp8s(w1[g]);
  sw[65536 + g]  = fp8s(w2[g]);
  sw[131072 + g] = fp8s(w3[g]);
}

__global__ __launch_bounds__(256) void k_dwconv(
    const float* __restrict__ x, const int8_t* __restrict__ swdw,
    const float* __restrict__ a1, float* __restrict__ t1,
    float* __restrict__ pS, float* __restrict__ pQ) {
  int n = blockIdx.x >> 8;
  int c = blockIdx.x & 255;
  const float* xp = x  + ((size_t)n * CCH + c) * HWSZ;
  float*       tp = t1 + ((size_t)n * CCH + c) * HWSZ;
  int wk[9];
#pragma unroll
  for (int i = 0; i < 9; ++i) wk[i] = (int)swdw[c * 9 + i];
  float slope = a1[c];
  float s = 0.f, q = 0.f;
  for (int p = threadIdx.x; p < HWSZ; p += 256) {
    int h = p / 56, w = p - h * 56;
    int acc = 0;
#pragma unroll
    for (int dh = -1; dh <= 1; ++dh) {
      int hh = h + dh;
      if (hh < 0 || hh >= 56) continue;
#pragma unroll
      for (int dw = -1; dw <= 1; ++dw) {
        int ww = w + dw;
        if (ww < 0 || ww >= 56) continue;
        float v = xp[hh * 56 + ww];
        int sv = (v > 0.f) - (v < 0.f);
        acc += sv * wk[(dh + 1) * 3 + (dw + 1)];
      }
    }
    float o = (float)acc;
    o = (o >= 0.f) ? o : slope * o;
    tp[p] = o;
    s += o; q += o * o;
  }
  __shared__ float rs[256], rq[256];
  rs[threadIdx.x] = s; rq[threadIdx.x] = q;
  __syncthreads();
  for (int off = 128; off > 0; off >>= 1) {
    if (threadIdx.x < off) {
      rs[threadIdx.x] += rs[threadIdx.x + off];
      rq[threadIdx.x] += rq[threadIdx.x + off];
    }
    __syncthreads();
  }
  if (threadIdx.x == 0) { pS[blockIdx.x] = rs[0]; pQ[blockIdx.x] = rq[0]; }
}

__global__ __launch_bounds__(256) void k_stats_dw(
    const float* __restrict__ pS, const float* __restrict__ pQ,
    const float* __restrict__ g, const float* __restrict__ b,
    float* __restrict__ scale, float* __restrict__ shift) {
  int c = threadIdx.x;
  float s = 0.f, q = 0.f;
  for (int n = 0; n < NIMG; ++n) { s += pS[n * CCH + c]; q += pQ[n * CCH + c]; }
  float mean = s / (float)MPIX;
  float var  = q / (float)MPIX - mean * mean;
  float inv  = rsqrtf(var + EPSBN);
  float sc   = g[c] * inv;
  scale[c] = sc;
  shift[c] = b[c] - mean * sc;
}

__global__ __launch_bounds__(256) void k_stats(
    const float* __restrict__ pS, const float* __restrict__ pQ, int nblocks,
    const float* __restrict__ g, const float* __restrict__ b,
    float* __restrict__ scale, float* __restrict__ shift) {
  int ch = blockIdx.x;
  int t = threadIdx.x;
  float s = 0.f, q = 0.f;
  for (int i = t; i < nblocks; i += 256) {
    s += pS[(size_t)i * 256 + ch];
    q += pQ[(size_t)i * 256 + ch];
  }
  __shared__ float rs[256], rq[256];
  rs[t] = s; rq[t] = q;
  __syncthreads();
  for (int off = 128; off > 0; off >>= 1) {
    if (t < off) { rs[t] += rs[t + off]; rq[t] += rq[t + off]; }
    __syncthreads();
  }
  if (t == 0) {
    float mean = rs[0] / (float)MPIX;
    float var  = rq[0] / (float)MPIX - mean * mean;
    float inv  = rsqrtf(var + EPSBN);
    float sc   = g[ch] * inv;
    scale[ch] = sc;
    shift[ch] = b[ch] - mean * sc;
  }
}

__global__ __launch_bounds__(256) void k_bn_res_pack(
    const float* __restrict__ t1, const float* __restrict__ x,
    const float* __restrict__ scale, const float* __restrict__ shift,
    float* __restrict__ r2, uint8_t* __restrict__ sa) {
  __shared__ float tile[32][33];
  int hwt = blockIdx.x % 98;
  int ct  = (blockIdx.x / 98) & 7;
  int n   = blockIdx.x / (98 * 8);
  int tx  = threadIdx.x & 31;
  int ty  = threadIdx.x >> 5;
#pragma unroll
  for (int j = 0; j < 4; ++j) {
    int cl = ty + j * 8;
    int c  = ct * 32 + cl;
    int p  = hwt * 32 + tx;
    size_t idx = ((size_t)n * CCH + c) * HWSZ + p;
    tile[cl][tx] = t1[idx] * scale[c] + shift[c] + x[idx];
  }
  __syncthreads();
#pragma unroll
  for (int j = 0; j < 4; ++j) {
    int pl = ty + j * 8;
    int cl = tx;
    size_t m = (size_t)n * HWSZ + hwt * 32 + pl;
    float v = tile[cl][pl];
    r2[m * CCH + ct * 32 + cl] = v;
    sa[m * CCH + ct * 32 + cl] = fp8s(v);
  }
}

__global__ __launch_bounds__(256) void k_gemm(
    const uint8_t* __restrict__ sa, const uint8_t* __restrict__ sw,
    const float* __restrict__ slope, float* __restrict__ tout,
    float* __restrict__ pS, float* __restrict__ pQ) {
  __shared__ float ps[256][4];
  __shared__ float pq[256][4];
  int t     = threadIdx.x;
  int wave  = t >> 5;
  int lane  = t & 31;
  int hi    = lane >> 4;
  int col16 = lane & 15;
  int mbase = blockIdx.x * 32 + (wave >> 2) * 16;
  int nbase = (wave & 3) * 64;
  int mrow  = mbase + col16;
  int sub   = (wave >> 2) * 2 + hi;

  v8f zero = {0.f, 0.f, 0.f, 0.f, 0.f, 0.f, 0.f, 0.f};
  v8f acc[4] = {zero, zero, zero, zero};

  const uint8_t* arow = sa + (size_t)mrow * CCH + hi * 8;
#pragma unroll
  for (int k0 = 0; k0 < CCH; k0 += 128) {
    v16i a;
    const uint8_t* ap = arow + k0;
#pragma unroll
    for (int u = 0; u < 8; ++u) {
      reinterpret_cast<int2*>(&a)[u] = *reinterpret_cast<const int2*>(ap + u * 16);
    }
#pragma unroll
    for (int nt = 0; nt < 4; ++nt) {
      int ch = nbase + nt * 16 + col16;
      const uint8_t* bp = sw + (size_t)ch * CCH + k0 + hi * 8;
      v16i b;
#pragma unroll
      for (int u = 0; u < 8; ++u) {
        reinterpret_cast<int2*>(&b)[u] = *reinterpret_cast<const int2*>(bp + u * 16);
      }
      acc[nt] = __builtin_amdgcn_wmma_f32_16x16x128_fp8_fp8(
          a, b, (short)0, acc[nt], false, false);
    }
  }

#pragma unroll
  for (int nt = 0; nt < 4; ++nt) {
    int ch = nbase + nt * 16 + col16;
    float al = slope[ch];
    float s = 0.f, q = 0.f;
    int mr = mbase + hi * 8;
#pragma unroll
    for (int j = 0; j < 8; ++j) {
      float v = acc[nt][j];
      v = (v >= 0.f) ? v : al * v;
      tout[(size_t)(mr + j) * CCH + ch] = v;
      s += v; q += v * v;
    }
    ps[ch][sub] = s;
    pq[ch][sub] = q;
  }
  __syncthreads();
  {
    float s = ps[t][0] + ps[t][1] + ps[t][2] + ps[t][3];
    float q = pq[t][0] + pq[t][1] + pq[t][2] + pq[t][3];
    pS[(size_t)blockIdx.x * 256 + t] = s;
    pQ[(size_t)blockIdx.x * 256 + t] = q;
  }
}

__global__ __launch_bounds__(256) void k_pack_elem(
    const float* __restrict__ tin, const float* __restrict__ res,
    const float* __restrict__ scale, const float* __restrict__ shift,
    uint8_t* __restrict__ sa, int hasres) {
  size_t i4 = (size_t)blockIdx.x * 256 + threadIdx.x;
  int ch0 = (int)((i4 * 4) & 255);
  float4 tv = reinterpret_cast<const float4*>(tin)[i4];
  float4 rv = {0.f, 0.f, 0.f, 0.f};
  if (hasres) rv = reinterpret_cast<const float4*>(res)[i4];
  float v0 = tv.x * scale[ch0 + 0] + shift[ch0 + 0] + rv.x;
  float v1 = tv.y * scale[ch0 + 1] + shift[ch0 + 1] + rv.y;
  float v2 = tv.z * scale[ch0 + 2] + shift[ch0 + 2] + rv.z;
  float v3 = tv.w * scale[ch0 + 3] + shift[ch0 + 3] + rv.w;
  uint32_t o = (uint32_t)fp8s(v0) | ((uint32_t)fp8s(v1) << 8) |
               ((uint32_t)fp8s(v2) << 16) | ((uint32_t)fp8s(v3) << 24);
  reinterpret_cast<uint32_t*>(sa)[i4] = o;
}

__global__ __launch_bounds__(256) void k_final(
    const float* __restrict__ tin, const float* __restrict__ scale,
    const float* __restrict__ shift, float* __restrict__ out) {
  __shared__ float tile[32][33];
  int hwt = blockIdx.x % 98;
  int ct  = (blockIdx.x / 98) & 7;
  int n   = blockIdx.x / (98 * 8);
  int tx  = threadIdx.x & 31;
  int ty  = threadIdx.x >> 5;
#pragma unroll
  for (int j = 0; j < 4; ++j) {
    int pl = ty + j * 8;
    int cl = tx;
    int c  = ct * 32 + cl;
    size_t m = (size_t)n * HWSZ + hwt * 32 + pl;
    tile[cl][pl] = tin[m * CCH + c] * scale[c] + shift[c];
  }
  __syncthreads();
#pragma unroll
  for (int j = 0; j < 4; ++j) {
    int cl = ty + j * 8;
    int c  = ct * 32 + cl;
    int p  = hwt * 32 + tx;
    out[((size_t)n * CCH + c) * HWSZ + p] = tile[cl][tx];
  }
}

extern "C" void kernel_launch(void* const* d_in, const int* in_sizes, int n_in,
                              void* d_out, int out_size, void* d_ws, size_t ws_size,
                              hipStream_t stream) {
  (void)in_sizes; (void)n_in; (void)out_size; (void)ws_size;
  const float* x   = (const float*)d_in[0];
  const float* wdw = (const float*)d_in[1];
  const float* w1  = (const float*)d_in[2];
  const float* w2  = (const float*)d_in[3];
  const float* w3  = (const float*)d_in[4];
  const float* a1  = (const float*)d_in[5];
  const float* a2  = (const float*)d_in[6];
  const float* a3  = (const float*)d_in[7];
  const float* a4  = (const float*)d_in[8];
  const float* g1  = (const float*)d_in[9];
  const float* b1  = (const float*)d_in[10];
  const float* g2  = (const float*)d_in[11];
  const float* b2  = (const float*)d_in[12];
  const float* g3  = (const float*)d_in[13];
  const float* b3  = (const float*)d_in[14];
  const float* g4  = (const float*)d_in[15];
  const float* b4  = (const float*)d_in[16];

  char* ws = (char*)d_ws;
  const size_t MC  = (size_t)MPIX * CCH;
  const size_t MC4 = MC * 4;
  float*   t1   = (float*)(ws);
  float*   r2   = (float*)(ws + MC4);
  float*   t2   = (float*)(ws + 2 * MC4);
  uint8_t* sa   = (uint8_t*)(ws + 3 * MC4);
  uint8_t* sw   = sa + MC;
  int8_t*  swdw = (int8_t*)(sw + 3 * 65536);
  float*   stats = (float*)(ws + 3 * MC4 + MC + 3 * 65536 + 4096);
  float*   pS   = stats + 2048;
  float*   pQ   = pS + (size_t)3136 * 256;

  float* sc1 = stats;        float* sh1 = stats + 256;
  float* sc2 = stats + 512;  float* sh2 = stats + 768;
  float* sc3 = stats + 1024; float* sh3 = stats + 1280;
  float* sc4 = stats + 1536; float* sh4 = stats + 1792;

  const int GEMM_BLOCKS = MPIX / 32;
  const int TR_BLOCKS   = NIMG * 8 * 98;
  const int EW_BLOCKS   = (int)(MC / 4 / 256);

  k_init<<<256, 256, 0, stream>>>(w1, w2, w3, wdw, sw, swdw);

  k_dwconv<<<NIMG * CCH, 256, 0, stream>>>(x, swdw, a1, t1, pS, pQ);
  k_stats_dw<<<1, 256, 0, stream>>>(pS, pQ, g1, b1, sc1, sh1);
  k_bn_res_pack<<<TR_BLOCKS, 256, 0, stream>>>(t1, x, sc1, sh1, r2, sa);

  k_gemm<<<GEMM_BLOCKS, 256, 0, stream>>>(sa, sw, a2, t2, pS, pQ);
  k_stats<<<256, 256, 0, stream>>>(pS, pQ, GEMM_BLOCKS, g2, b2, sc2, sh2);
  k_pack_elem<<<EW_BLOCKS, 256, 0, stream>>>(t2, r2, sc2, sh2, sa, 1);

  k_gemm<<<GEMM_BLOCKS, 256, 0, stream>>>(sa, sw + 65536, a3, t1, pS, pQ);
  k_stats<<<256, 256, 0, stream>>>(pS, pQ, GEMM_BLOCKS, g3, b3, sc3, sh3);
  k_pack_elem<<<EW_BLOCKS, 256, 0, stream>>>(t1, nullptr, sc3, sh3, sa, 0);

  k_gemm<<<GEMM_BLOCKS, 256, 0, stream>>>(sa, sw + 131072, a4, t2, pS, pQ);
  k_stats<<<256, 256, 0, stream>>>(pS, pQ, GEMM_BLOCKS, g4, b4, sc4, sh4);
  k_final<<<TR_BLOCKS, 256, 0, stream>>>(t2, sc4, sh4, (float*)d_out);
}
